// CFD_DC_19447611916759
// MI455X (gfx1250) — matrix operand order repaired and checked by reading, not compiled, not run on hardware
//
#include <hip/hip_runtime.h>
#include <hip/hip_bf16.h>
#include <math.h>

// ---------------- problem constants ----------------
#define BB   4096
#define NN   6
#define FF   1024
#define FOO  512
#define CC   100
#define HIDD 768
#define REDD 3
#define DD   (FF + (NN-1)*FOO)   // 3584
#define EPSV 1e-5f

typedef _Float16 half_t;
typedef __attribute__((ext_vector_type(16))) _Float16 v16h;
typedef __attribute__((ext_vector_type(8)))  _Float16 v8h;
typedef __attribute__((ext_vector_type(8)))  float    v8f;

// ---------------- helpers ----------------
__device__ __forceinline__ v8h ld8h(const half_t* p) { return *(const v8h*)p; }

// ---------------- f32 -> f16 convert ----------------
__global__ void cvt_f32_f16(const float* __restrict__ src,
                            half_t* __restrict__ dst, long n) {
  long i = (long)blockIdx.x * blockDim.x + threadIdx.x;
  long stride = (long)gridDim.x * blockDim.x;
  for (; i < n; i += stride) dst[i] = (half_t)src[i];
}

// ---------------- fold BN into scale/shift ----------------
// y_bn = ((gemm + b) - rm) * g*rsqrt(rv+eps) + be  ==  gemm*s + t
__global__ void fuse_bn(const float* __restrict__ b, const float* __restrict__ g,
                        const float* __restrict__ be, const float* __restrict__ rm,
                        const float* __restrict__ rv,
                        float* __restrict__ s, float* __restrict__ t, int n) {
  int i = blockIdx.x * blockDim.x + threadIdx.x;
  if (i >= n) return;
  float sc = g[i] * rsqrtf(rv[i] + EPSV);
  s[i] = sc;
  t[i] = (b[i] - rm[i]) * sc + be[i];
}

// ---------------- wave-per-row mean over f16 rows ----------------
__global__ void row_mean_f16(const half_t* __restrict__ src, int rows, int cols,
                             int stride, float* __restrict__ out) {
  int row = blockIdx.x * (blockDim.x >> 5) + (threadIdx.x >> 5);
  if (row >= rows) return;
  int lane = threadIdx.x & 31;
  const half_t* p = src + (size_t)row * stride;
  float s = 0.f;
  for (int c = lane; c < cols; c += 32) s += (float)p[c];
  #pragma unroll
  for (int off = 16; off > 0; off >>= 1) s += __shfl_xor(s, off, 32);
  if (lane == 0) out[row] = s / (float)cols;
}

// ---------------- NT GEMM: C[M x Ncols] = A[M x K] * B[Ncols x K]^T ----------------
// A row-major (lda), B row-major over [Ncols][K] (ldb).  f16 in, f32 WMMA accum.
// Wave tile 64x32 (4x2 WMMA tiles); workgroup 8 waves -> 256x64 macro tile.
// K is a compile-time constant -> fully unrolled, loads use immediate offsets.
// Explicit double buffering: next step's 12 loads issue before current WMMAs.
// Out-of-range B columns are CLAMPED: garbage lands only in columns never stored.
// EPI 0: out_f16 = relu(acc*scale[col] + shift[col])
// EPI 1: out_f32 = acc   (raw partials)

#define LOAD_FRAGS(AF, BF, KK) do {                                     \
    _Pragma("unroll")                                                   \
    for (int mt = 0; mt < 4; ++mt) {                                    \
      v8h lo = ld8h(Ab + aoff[mt] + (KK));                              \
      v8h hi = ld8h(Ab + aoff[mt] + (KK) + 16);                         \
      AF[mt] = __builtin_shufflevector(lo, hi,                          \
                 0,1,2,3,4,5,6,7,8,9,10,11,12,13,14,15);                \
    }                                                                   \
    _Pragma("unroll")                                                   \
    for (int nt = 0; nt < 2; ++nt) {                                    \
      v8h lo = ld8h(Bb + boff[nt] + (KK));                              \
      v8h hi = ld8h(Bb + boff[nt] + (KK) + 16);                         \
      BF[nt] = __builtin_shufflevector(lo, hi,                          \
                 0,1,2,3,4,5,6,7,8,9,10,11,12,13,14,15);                \
    }                                                                   \
  } while (0)

#define DO_WMMAS(AF, BF) do {                                           \
    _Pragma("unroll")                                                   \
    for (int mt = 0; mt < 4; ++mt)                                      \
      _Pragma("unroll")                                                 \
      for (int nt = 0; nt < 2; ++nt)                                    \
        acc[mt][nt] = __builtin_amdgcn_wmma_f32_16x16x32_f16(           \
            false, AF[mt], false, BF[nt], (short)0, acc[mt][nt],        \
            false, false);                                              \
  } while (0)

template<int EPI, int K>
__global__ __launch_bounds__(256)
void gemm_nt(const half_t* __restrict__ Ab, int lda,
             const half_t* __restrict__ Bb, int ldb,
             int Ncols,
             const float* __restrict__ scale, const float* __restrict__ shift,
             void* __restrict__ outp, int ldc) {
  const int lane = threadIdx.x & 31;
  const int wave = threadIdx.x >> 5;
  const int half = lane >> 4;     // which 16-lane group
  const int lrow = lane & 15;
  const int m_base = blockIdx.x * 256 + (wave >> 1) * 64;
  const int n_base = blockIdx.y * 64  + (wave & 1) * 32;

  v8f acc[4][2] = {};

  // 32-bit element offsets from uniform bases (saddr + voffset addressing).
  // A lane mapping: row = m0 + lrow; lane holds K = half*8..+7 and 16+half*8..+7
  int aoff[4];
  #pragma unroll
  for (int mt = 0; mt < 4; ++mt)
    aoff[mt] = (m_base + mt * 16 + lrow) * lda + half * 8;
  // B lane mapping: column = n0 + lrow (clamped); lane holds K = half*8..+7 and 16+half*8..+7
  int colIdx[2], boff[2];
  #pragma unroll
  for (int nt = 0; nt < 2; ++nt) {
    int c = n_base + nt * 16 + lrow;
    colIdx[nt] = c;
    int cl = (c < Ncols) ? c : (Ncols - 1);
    boff[nt] = cl * ldb + half * 8;
  }

  v16h a0[4], b0[2], a1[4], b1[2];
  LOAD_FRAGS(a0, b0, 0);
  #pragma unroll
  for (int k0 = 0; k0 < K; k0 += 64) {
    LOAD_FRAGS(a1, b1, k0 + 32);
    DO_WMMAS(a0, b0);
    if (k0 + 64 < K) LOAD_FRAGS(a0, b0, k0 + 64);
    DO_WMMAS(a1, b1);
  }

  // D layout: vgpr r, lanes 0-15 -> (M=r, N=lane); lanes 16-31 -> (M=8+r, N=lane-16)
  #pragma unroll
  for (int nt = 0; nt < 2; ++nt) {
    int col = colIdx[nt];
    if (col >= Ncols) continue;
    float sc = 1.f, sh = 0.f;
    if (EPI == 0) { sc = scale[col]; sh = shift[col]; }
    #pragma unroll
    for (int mt = 0; mt < 4; ++mt) {
      int rowb = m_base + mt * 16 + half * 8;
      #pragma unroll
      for (int r = 0; r < 8; ++r) {
        float v = acc[mt][nt][r];
        if (EPI == 0) {
          v = v * sc + sh;
          v = v > 0.f ? v : 0.f;
          ((half_t*)outp)[(size_t)(rowb + r) * ldc + col] = (half_t)v;
        } else {
          ((float*)outp)[(size_t)(rowb + r) * ldc + col] = v;
        }
      }
    }
  }
}

// ---------------- gating MLP: pool(6) -> relu(3) -> sigmoid(6) ----------------
__global__ void gate_kernel(const float* __restrict__ xmean,
                            const float* __restrict__ cmean,
                            const float* __restrict__ Wa,  // (N, RED, N)
                            const float* __restrict__ Wb,  // (N, N, RED)
                            float* __restrict__ w, int total /* B*N */) {
  int i = blockIdx.x * blockDim.x + threadIdx.x;
  if (i >= total) return;
  int n = i % NN;
  int b = i / NN;
  float pool[NN];
  pool[0] = xmean[i];
  #pragma unroll
  for (int j = 0; j < NN - 1; ++j) {
    int m = (j < n) ? j : j + 1;
    pool[1 + j] = cmean[b * NN + m];
  }
  float r[REDD];
  #pragma unroll
  for (int k = 0; k < REDD; ++k) {
    float a = 0.f;
    #pragma unroll
    for (int t = 0; t < NN; ++t) a += Wa[(n * REDD + k) * NN + t] * pool[t];
    r[k] = a > 0.f ? a : 0.f;
  }
  #pragma unroll
  for (int m = 0; m < NN; ++m) {
    float a = 0.f;
    #pragma unroll
    for (int k = 0; k < REDD; ++k) a += Wb[(n * NN + m) * REDD + k] * r[k];
    w[i * NN + m] = 1.f / (1.f + expf(-a));
  }
}

// ---------------- combine: out = bc + sum_s w_s * P_s ----------------
__global__ void combine_kernel(const float* __restrict__ P,
                               const float* __restrict__ w,
                               const float* __restrict__ bc,
                               float* __restrict__ out, long total /* B*N*C */) {
  long i = (long)blockIdx.x * blockDim.x + threadIdx.x;
  if (i >= total) return;
  int  c  = (int)(i % CC);
  long bn = i / CC;
  int  n  = (int)(bn % NN);
  float acc = bc[n * CC + c];
  #pragma unroll
  for (int s = 0; s < NN; ++s)
    acc += w[bn * NN + s] * P[(size_t)s * total + i];
  out[i] = acc;
}

// ---------------- host side ----------------
extern "C" void kernel_launch(void* const* d_in, const int* in_sizes, int n_in,
                              void* d_out, int out_size, void* d_ws, size_t ws_size,
                              hipStream_t stream) {
  (void)in_sizes; (void)n_in; (void)out_size; (void)ws_size;
  const float* x   = (const float*)d_in[0];
  const float* W1  = (const float*)d_in[1];
  const float* b1  = (const float*)d_in[2];
  const float* g1  = (const float*)d_in[3];
  const float* be1 = (const float*)d_in[4];
  const float* rm1 = (const float*)d_in[5];
  const float* rv1 = (const float*)d_in[6];
  const float* W2  = (const float*)d_in[7];
  const float* b2  = (const float*)d_in[8];
  const float* g2  = (const float*)d_in[9];
  const float* be2 = (const float*)d_in[10];
  const float* rm2 = (const float*)d_in[11];
  const float* rv2 = (const float*)d_in[12];
  const float* Wa  = (const float*)d_in[13];
  const float* Wb  = (const float*)d_in[14];
  const float* Wc  = (const float*)d_in[15];
  const float* bc  = (const float*)d_in[16];
  float* out = (float*)d_out;

  // workspace carve-up (256B aligned)
  char* ws = (char*)d_ws;
  size_t off = 0;
  auto take = [&](size_t bytes) -> char* {
    char* p = ws + off;
    off += (bytes + 255) & ~(size_t)255;
    return p;
  };
  half_t* xb   = (half_t*)take((size_t)BB * NN * FF * 2);
  half_t* W1h  = (half_t*)take((size_t)NN * HIDD * FF * 2);
  half_t* W2h  = (half_t*)take((size_t)NN * FOO * HIDD * 2);
  half_t* Wch  = (half_t*)take((size_t)NN * CC * DD * 2);
  half_t* hb   = (half_t*)take((size_t)BB * NN * HIDD * 2);
  half_t* compb= (half_t*)take((size_t)BB * NN * FOO * 2);
  float*  s1   = (float*) take((size_t)NN * HIDD * 4);
  float*  t1   = (float*) take((size_t)NN * HIDD * 4);
  float*  s2   = (float*) take((size_t)NN * FOO * 4);
  float*  t2   = (float*) take((size_t)NN * FOO * 4);
  float*  xmean= (float*) take((size_t)BB * NN * 4);
  float*  cmean= (float*) take((size_t)BB * NN * 4);
  float*  wbuf = (float*) take((size_t)BB * NN * NN * 4);
  long    BNC  = (long)BB * NN * CC;
  float*  Pbuf = (float*) take((size_t)NN * BNC * 4);

  // 1) convert to f16
  cvt_f32_f16<<<4096, 256, 0, stream>>>(x,  xb,  (long)BB * NN * FF);
  cvt_f32_f16<<<2048, 256, 0, stream>>>(W1, W1h, (long)NN * HIDD * FF);
  cvt_f32_f16<<<2048, 256, 0, stream>>>(W2, W2h, (long)NN * FOO * HIDD);
  cvt_f32_f16<<<2048, 256, 0, stream>>>(Wc, Wch, (long)NN * CC * DD);

  // 2) fold BN params
  fuse_bn<<<(NN * HIDD + 255) / 256, 256, 0, stream>>>(b1, g1, be1, rm1, rv1, s1, t1, NN * HIDD);
  fuse_bn<<<(NN * FOO  + 255) / 256, 256, 0, stream>>>(b2, g2, be2, rm2, rv2, s2, t2, NN * FOO);

  // 3) x row means (wave per row)
  row_mean_f16<<<(BB * NN + 7) / 8, 256, 0, stream>>>(xb, BB * NN, FF, FF, xmean);

  // 4) stage-1 GEMMs: h = relu(bn(x @ W1^T)), K = FF
  for (int n = 0; n < NN; ++n) {
    gemm_nt<0, FF><<<dim3(BB / 256, HIDD / 64), 256, 0, stream>>>(
        xb + n * FF, NN * FF,
        W1h + (size_t)n * HIDD * FF, FF,
        HIDD,
        s1 + n * HIDD, t1 + n * HIDD,
        hb + n * HIDD, NN * HIDD);
  }

  // 5) stage-2 GEMMs: comp = relu(bn(h @ W2^T)), K = HIDD
  for (int n = 0; n < NN; ++n) {
    gemm_nt<0, HIDD><<<dim3(BB / 256, FOO / 64), 256, 0, stream>>>(
        hb + n * HIDD, NN * HIDD,
        W2h + (size_t)n * FOO * HIDD, HIDD,
        FOO,
        s2 + n * FOO, t2 + n * FOO,
        compb + n * FOO, NN * FOO);
  }

  // 6) comp row means + gating weights
  row_mean_f16<<<(BB * NN + 7) / 8, 256, 0, stream>>>(compb, BB * NN, FOO, FOO, cmean);
  gate_kernel<<<(BB * NN + 255) / 256, 256, 0, stream>>>(xmean, cmean, Wa, Wb, wbuf, BB * NN);

  // 7) final partial GEMMs: P[s][b,n,c] = seg_s(feat) @ Wc_seg^T
  for (int n = 0; n < NN; ++n) {
    for (int s = 0; s < NN; ++s) {
      float* Pp = Pbuf + (size_t)s * BNC + n * CC;
      if (s == 0) {
        gemm_nt<1, FF><<<dim3(BB / 256, (CC + 63) / 64), 256, 0, stream>>>(
            xb + n * FF, NN * FF,
            Wch + (size_t)n * CC * DD + 0, DD,
            CC, nullptr, nullptr, Pp, NN * CC);
      } else {
        int m = (s - 1 < n) ? (s - 1) : s;      // IDX[n][s-1]
        gemm_nt<1, FOO><<<dim3(BB / 256, (CC + 63) / 64), 256, 0, stream>>>(
            compb + m * FOO, NN * FOO,
            Wch + (size_t)n * CC * DD + FF + (s - 1) * FOO, DD,
            CC, nullptr, nullptr, Pp, NN * CC);
      }
    }
  }

  // 8) weighted combine + bias
  combine_kernel<<<(int)((BNC + 255) / 256), 256, 0, stream>>>(Pbuf, wbuf, bc, out, BNC);
}
